// RealISTFT_89833535963939
// MI455X (gfx1250) — matrix operand order repaired and checked by reading, not compiled, not run on hardware
//
#include <hip/hip_runtime.h>
#include <hip/hip_bf16.h>
#include <math.h>

typedef __attribute__((ext_vector_type(16))) __bf16        v16bf;
typedef __attribute__((ext_vector_type(8)))  float         v8f;
typedef __attribute__((ext_vector_type(4)))  unsigned int  u32x4;

union FragBF { u32x4 u[2]; v16bf v; };

#define N_FFT     1024
#define HOP       256
#define N_FREQ    513
#define BATCH     16
#define T_FRAMES  4000
#define KSTEPS    17                       // ceil(513/32), K padded to 544
#define NTILES    64                       // 1024/16
#define MBLK      32                       // frames per workgroup (2 M-tiles/wave)
#define MT_PER_B  (T_FRAMES / MBLK)        // 125
#define OUT_PER_B ((T_FRAMES - 1) * HOP)   // 1,023,744
#define BASIS_HALF_COUNT (KSTEPS * NTILES * 2 * 32 * 16)
#define BASIS_BYTES      (BASIS_HALF_COUNT * 2)
#define ROWH 40                            // LDS row stride in halves (80B, b128-aligned)

__device__ __forceinline__ unsigned short f2bf(float f) {
  unsigned int u = __float_as_uint(f);
  u += 0x7FFFu + ((u >> 16) & 1u);         // round-to-nearest-even
  return (unsigned short)(u >> 16);
}
__device__ __forceinline__ float bf2f(unsigned short h) {
  return __uint_as_float(((unsigned int)h) << 16);
}

// ---------------------------------------------------------------------------
// Kernel 1: window-folded synthesis basis, pre-swizzled into WMMA B-fragment
// order: [ks][ntile][cs][lane][half]; lane&15 = N col, K = (lane>>4)*16 + half.
// cs=0 -> cos*scale*win, cs=1 -> -sin*scale*win. f >= 513 zero-padded.
// ---------------------------------------------------------------------------
__global__ void basis_kernel(const float* __restrict__ window,
                             unsigned short* __restrict__ basis) {
  int idx = blockIdx.x * 256 + threadIdx.x;
  if (idx >= BASIS_HALF_COUNT) return;
  int h    = idx & 15;
  int lane = (idx >> 4) & 31;
  int cs   = (idx >> 9) & 1;
  int nt   = (idx >> 10) & 63;
  int ks   = idx >> 16;

  int n = nt * 16 + (lane & 15);
  int f = ks * 32 + ((lane >> 4) << 4) + h;

  float val = 0.0f;
  if (f < N_FREQ) {
    float scale = (f == 0 || f == N_FFT / 2) ? (1.0f / N_FFT) : (2.0f / N_FFT);
    unsigned int ph = ((unsigned int)n * (unsigned int)f) & (N_FFT - 1); // exact reduction
    float ang = (float)ph * (6.28318530717958647692f / (float)N_FFT);
    float base = (cs == 0) ? __cosf(ang) : -__sinf(ang);
    val = base * scale * window[n];
  }
  basis[idx] = f2bf(val);
}

// ---------------------------------------------------------------------------
// Kernel 2: fused S=mag*phase staging + bf16 WMMA GEMM, 32 frames x 1024 n
// per 512-thread block. Wave w: M-tiles {0,1}, N-tiles [4w, 4w+4).
// Ping-pong LDS staging, one barrier per K-step, staging loads issued before
// B loads so in-order load completion hides HBM latency behind 16 WMMAs.
// ---------------------------------------------------------------------------
__global__ __launch_bounds__(512) void gemm_kernel(
    const float* __restrict__ mag, const float* __restrict__ cosp,
    const float* __restrict__ sinp, const unsigned short* __restrict__ basis,
    unsigned short* __restrict__ frames) {
  __shared__ __align__(16) unsigned short ArS[2][MBLK * ROWH];
  __shared__ __align__(16) unsigned short AiS[2][MBLK * ROWH];

  const int blk  = blockIdx.x;
  const int b    = blk / MT_PER_B;
  const int t0   = (blk % MT_PER_B) * MBLK;
  const int tid  = threadIdx.x;
  const int lane = tid & 31;
  const int w    = tid >> 5;

  // staging coords: 512 threads x 2 elems cover 32 t x 32 f
  const int tloc  = tid & 31;        // 0..31 (contiguous t -> coalesced)
  const int fbase = tid >> 5;        // 0..15 ; f' = fbase + 16*e
  const size_t inCol = ((size_t)b * N_FREQ) * T_FRAMES + (size_t)(t0 + tloc);

  v8f acc[8] = {};                   // [mt*4 + nt]

  const int row = lane & 15;
  const int sel = lane >> 4;

  // prologue: stage ks = 0 into buffer 0
#pragma unroll
  for (int e = 0; e < 2; ++e) {
    int fp = fbase + 16 * e;         // == global f for ks=0
    float ar = 0.0f, ai = 0.0f;
    if (fp < N_FREQ) {
      size_t ix = inCol + (size_t)fp * T_FRAMES;
      float m = mag[ix];
      ar = m * cosp[ix];
      ai = m * sinp[ix];
    }
    ArS[0][tloc * ROWH + fp] = f2bf(ar);
    AiS[0][tloc * ROWH + fp] = f2bf(ai);
  }
  __syncthreads();

  const u32x4* bp = (const u32x4*)basis;

  for (int ks = 0; ks < KSTEPS; ++ks) {
    const int cur = ks & 1, nxt = cur ^ 1;

    // A fragments for both M-tiles (contiguous K order: K = (lane>>4)*16 + half, matches basis)
    FragBF aR[2], aI[2];
#pragma unroll
    for (int mt = 0; mt < 2; ++mt) {
      const unsigned short* pr = &ArS[cur][(mt * 16 + row) * ROWH + sel * 16];
      const unsigned short* pi = &AiS[cur][(mt * 16 + row) * ROWH + sel * 16];
      aR[mt].u[0] = *(const u32x4*)pr;
      aR[mt].u[1] = *(const u32x4*)(pr + 8);
      aI[mt].u[0] = *(const u32x4*)pi;
      aI[mt].u[1] = *(const u32x4*)(pi + 8);
    }

    // prefetch next K-step staging data (HBM) early; consumed after WMMAs
    const bool donext = (ks + 1 < KSTEPS);
    float sm[2] = {0.0f, 0.0f}, sc[2] = {0.0f, 0.0f}, ssn[2] = {0.0f, 0.0f};
#pragma unroll
    for (int e = 0; e < 2; ++e) {
      int f = (ks + 1) * 32 + fbase + 16 * e;
      if (donext && f < N_FREQ) {
        size_t ix = inCol + (size_t)f * T_FRAMES;
        sm[e]  = mag[ix];
        sc[e]  = cosp[ix];
        ssn[e] = sinp[ix];
      }
    }

    // B fragments (L2-resident basis) + 16 WMMAs
#pragma unroll
    for (int nt = 0; nt < 4; ++nt) {
      size_t i0 = ((((size_t)ks * NTILES + (w * 4 + nt)) * 2 + 0) * 32 + lane) * 2;
      FragBF bC, bS;
      bC.u[0] = bp[i0];      bC.u[1] = bp[i0 + 1];
      bS.u[0] = bp[i0 + 64]; bS.u[1] = bp[i0 + 65];   // cs=1 block: +32*2 uint4
#pragma unroll
      for (int mt = 0; mt < 2; ++mt) {
        acc[mt * 4 + nt] = __builtin_amdgcn_wmma_f32_16x16x32_bf16(
            false, aR[mt].v, false, bC.v, (short)0, acc[mt * 4 + nt], false, false);
        acc[mt * 4 + nt] = __builtin_amdgcn_wmma_f32_16x16x32_bf16(
            false, aI[mt].v, false, bS.v, (short)0, acc[mt * 4 + nt], false, false);
      }
    }

    // convert + store staged tile into the other buffer
    if (donext) {
#pragma unroll
      for (int e = 0; e < 2; ++e) {
        int fp = fbase + 16 * e;
        ArS[nxt][tloc * ROWH + fp] = f2bf(sm[e] * sc[e]);
        AiS[nxt][tloc * ROWH + fp] = f2bf(sm[e] * ssn[e]);
      }
    }
    __syncthreads();
  }

  // Epilogue: lane -> N col, VGPR r -> M row (+8 for hi lanes); window already
  // folded into basis, so store windowed frames directly as bf16.
#pragma unroll
  for (int mt = 0; mt < 2; ++mt) {
    const int trowbase = t0 + mt * 16 + sel * 8;
#pragma unroll
    for (int nt = 0; nt < 4; ++nt) {
      v8f a = acc[mt * 4 + nt];
      int n0 = (w * 4 + nt) * 16 + row;
#pragma unroll
      for (int r = 0; r < 8; ++r) {
        size_t o = ((size_t)(b * T_FRAMES + trowbase + r)) * N_FFT + n0;
        frames[o] = f2bf(a[r]);
      }
    }
  }
}

// ---------------------------------------------------------------------------
// Kernel 3: gather overlap-add + exact edge envelope + trim. No atomics.
// ---------------------------------------------------------------------------
__global__ void ola_kernel(const unsigned short* __restrict__ frames,
                           const float* __restrict__ window,
                           float* __restrict__ out, int total) {
  int idx = blockIdx.x * 256 + threadIdx.x;
  if (idx >= total) return;
  int b  = idx / OUT_PER_B;
  int sp = idx - b * OUT_PER_B;
  int s  = sp + N_FFT / 2;               // un-trimmed sample index
  int thi = s >> 8;                      // s / HOP
  float acc = 0.0f, env = 0.0f;
#pragma unroll
  for (int dt = 0; dt < 4; ++dt) {
    int t = thi - dt;
    if (t >= 0 && t < T_FRAMES) {
      int n = s - (t << 8);              // in [0, 1024)
      float wv = window[n];
      env += wv * wv;
      acc += bf2f(frames[((size_t)b * T_FRAMES + t) * N_FFT + n]);
    }
  }
  out[idx] = acc / (env + 1e-11f);
}

// ---------------------------------------------------------------------------
extern "C" void kernel_launch(void* const* d_in, const int* in_sizes, int n_in,
                              void* d_out, int out_size, void* d_ws, size_t ws_size,
                              hipStream_t stream) {
  const float* mag  = (const float*)d_in[0];
  const float* cosp = (const float*)d_in[1];
  const float* sinp = (const float*)d_in[2];
  const float* win  = (const float*)d_in[3];
  float* out = (float*)d_out;

  unsigned short* basis  = (unsigned short*)d_ws;
  unsigned short* frames = (unsigned short*)((char*)d_ws + BASIS_BYTES);

  basis_kernel<<<(BASIS_HALF_COUNT + 255) / 256, 256, 0, stream>>>(win, basis);
  gemm_kernel<<<BATCH * MT_PER_B, 512, 0, stream>>>(mag, cosp, sinp, basis, frames);
  ola_kernel<<<(out_size + 255) / 256, 256, 0, stream>>>(frames, win, out, out_size);
}
